// STSENC_8340826489519
// MI455X (gfx1250) — matrix operand order repaired and checked by reading, not compiled, not run on hardware
//
#include <hip/hip_runtime.h>
#include <cstdint>
#include <cstddef>

#define BN_EPS   1e-5f
#define NB_STATS 240

typedef __attribute__((ext_vector_type(16))) __bf16         v16bf;
typedef __attribute__((ext_vector_type(8)))  float          v8f;
typedef __attribute__((ext_vector_type(16))) unsigned short us16;
typedef __attribute__((ext_vector_type(8)))  unsigned short us8;

__device__ __forceinline__ unsigned short f2bf(float f) {
  unsigned u = __builtin_bit_cast(unsigned, f);
  u += 0x7FFFu + ((u >> 16) & 1u);          // round-to-nearest-even
  return (unsigned short)(u >> 16);
}

// ---------------- small elementwise kernels ----------------
__global__ void k_silu(const float* __restrict__ t, float* __restrict__ o, int n) {
  int i = blockIdx.x * blockDim.x + threadIdx.x;
  if (i < n) { float v = t[i]; o[i] = v / (1.f + __expf(-v)); }
}

// convert fp32 weights (rows x cols) -> zero-padded bf16 (Rpad x Kpad)
__global__ void k_wconv(const float* __restrict__ src, unsigned short* __restrict__ dst,
                        int rows, int cols, int Kpad, long long total) {
  long long i = (long long)blockIdx.x * blockDim.x + threadIdx.x;
  if (i >= total) return;
  int k = (int)(i % Kpad); int r = (int)(i / Kpad);
  dst[i] = (r < rows && k < cols) ? f2bf(src[(long long)r * cols + k]) : (unsigned short)0;
}

// x (N,2,12,17) fp32 -> pixel-major fp32 (P x 2) and bf16 (P x 32, zero-padded)
__global__ void k_init(const float* __restrict__ x, float* __restrict__ xf,
                       unsigned short* __restrict__ xb, int N) {
  long long total = (long long)N * 12 * 17 * 32;
  long long i = (long long)blockIdx.x * blockDim.x + threadIdx.x;
  if (i >= total) return;
  int c = (int)(i % 32); long long p = i / 32;
  int v = (int)(p % 17); int t = (int)((p / 17) % 12); int n = (int)(p / (17 * 12));
  if (c < 2) {
    float val = x[(((long long)n * 2 + c) * 12 + t) * 17 + v];
    xf[p * 2 + c] = val;
    xb[i] = f2bf(val);
  } else {
    xb[i] = 0;
  }
}

// fused temporal (T) + spatial (A) graph mix, emits bf16 GEMM operand (P x Kpad)
// x: pixel-major (N*12*V, Cin) fp32;  Tm: (V,12,12);  Am: (12,V,V)
__global__ void k_mix(const float* __restrict__ x, unsigned short* __restrict__ h2,
                      const float* __restrict__ Tm, const float* __restrict__ Am,
                      int N, int V, int Cin, int Kpad) {
  long long total = (long long)N * 12 * Kpad;
  long long i = (long long)blockIdx.x * blockDim.x + threadIdx.x;
  if (i >= total) return;
  int c = (int)(i % Kpad); long long r = i / Kpad;
  int q = (int)(r % 12); int n = (int)(r / 12);
  unsigned short* out = h2 + ((long long)(n * 12 + q) * V) * Kpad + c;
  if (c >= Cin) { for (int w = 0; w < V; ++w) out[(long long)w * Kpad] = 0; return; }
  float tmp[17];
  for (int v = 0; v < V; ++v) {
    float s = 0.f;
    const float* xp = x + (((long long)(n * 12)) * V + v) * Cin + c;   // t = 0
    for (int t = 0; t < 12; ++t) s += xp[(long long)t * V * Cin] * Tm[(v * 12 + t) * 12 + q];
    tmp[v] = s;
  }
  for (int w = 0; w < V; ++w) {
    float s = 0.f;
    for (int v = 0; v < V; ++v) s += tmp[v] * Am[(q * V + v) * V + w];
    out[(long long)w * Kpad] = f2bf(s);
  }
}

// ---------------- WMMA bf16 GEMM ----------------
// D[p, o] = bias[o] + sum_k A[o, k] * B[p, k]
// A: (Mpad x K) bf16 row-major (Mpad = 16*gridDim.y), B: (P x K) bf16 row-major,
// D: (P x O) fp32. K % 32 == 0, P % 256 == 0. Block = 128 (4 waves), wave -> 16x64 tile.
__global__ void __launch_bounds__(128)
k_gemm(const unsigned short* __restrict__ A, const unsigned short* __restrict__ B,
       float* __restrict__ D, const float* __restrict__ bias,
       int O, int K, long long P) {
  int lane = threadIdx.x & 31;
  int wave = threadIdx.x >> 5;
  int half = lane >> 4;          // 0: lanes 0-15, 1: lanes 16-31
  int lm   = lane & 15;
  long long p0 = ((long long)blockIdx.x * 4 + wave) * 64;
  int m0 = blockIdx.y * 16;

  v8f acc[4];
  for (int j = 0; j < 4; ++j)
    for (int r = 0; r < 8; ++r) {
      int o = m0 + r + half * 8;
      acc[j][r] = (o < O) ? bias[o] : 0.0f;
    }

  for (int k0 = 0; k0 < K; k0 += 32) {
    // A fragment: row m = m0+lm; lanes 0-15 hold K k0..k0+15, lanes 16-31 k0+16..k0+31
    const unsigned short* arow = A + (size_t)(m0 + lm) * K + k0 + half * 16;
    us8 alo = *(const us8*)(arow);
    us8 ahi = *(const us8*)(arow + 8);
    us16 av = __builtin_shufflevector(alo, ahi, 0,1,2,3,4,5,6,7,8,9,10,11,12,13,14,15);
    v16bf af = __builtin_bit_cast(v16bf, av);
    for (int j = 0; j < 4; ++j) {
      long long p = p0 + j * 16 + lm;
      // B fragment: column n = lm; lanes 0-15 hold K k0..k0+15, lanes 16-31 k0+16..k0+31
      const unsigned short* brow = B + (size_t)p * K + k0 + half * 16;
      us16 bv = *(const us16*)brow;
      v16bf bfv = __builtin_bit_cast(v16bf, bv);
      acc[j] = __builtin_amdgcn_wmma_f32_16x16x32_bf16(false, af, false, bfv,
                                                       (short)0, acc[j], false, false);
    }
  }
  for (int j = 0; j < 4; ++j) {
    long long p = p0 + j * 16 + lm;
    for (int r = 0; r < 8; ++r) {
      int o = m0 + r + half * 8;
      if (o < O) D[(size_t)p * O + o] = acc[j][r];
    }
  }
}

// ---------------- BN statistics ----------------
// channel(i) = (i / inner) % nch ; per-block partial sums via LDS atomics (nch <= 128)
__global__ void k_stats(const float* __restrict__ src, long long total, int inner, int nch,
                        float* __restrict__ part) {
  __shared__ float s[256];
  for (int i = threadIdx.x; i < nch * 2; i += blockDim.x) s[i] = 0.f;
  __syncthreads();
  long long stride = (long long)gridDim.x * blockDim.x;
  for (long long i = (long long)blockIdx.x * blockDim.x + threadIdx.x; i < total; i += stride) {
    float v = src[i];
    int ch = (int)((i / inner) % nch);
    atomicAdd(&s[ch * 2], v);
    atomicAdd(&s[ch * 2 + 1], v * v);
  }
  __syncthreads();
  for (int i = threadIdx.x; i < nch * 2; i += blockDim.x)
    part[(long long)blockIdx.x * nch * 2 + i] = s[i];
}

__global__ void k_sreduce(const float* __restrict__ part, int nb, int nch,
                          float* __restrict__ st) {
  int i = threadIdx.x;
  if (i < nch * 2) {
    float s = 0.f;
    for (int b = 0; b < nb; ++b) s += part[(long long)b * nch * 2 + i];
    st[i] = s;
  }
}

// emb[n,o] = eb[o] + sum_j silu_t[n,j] * ew[o,j]
__global__ void k_emb(const float* __restrict__ semb, const float* __restrict__ ew,
                      const float* __restrict__ eb, float* __restrict__ emb, int N, int O) {
  long long total = (long long)N * O;
  long long i = (long long)blockIdx.x * blockDim.x + threadIdx.x;
  if (i >= total) return;
  int o = (int)(i % O); int n = (int)(i / O);
  float s = eb[o];
  for (int j = 0; j < 12; ++j) s += semb[n * 12 + j] * ew[o * 12 + j];
  emb[i] = s;
}

// BN + residual + PReLU + emb; in-place fp32 y over hc; optional bf16 shadow out
__global__ void k_finish(float* __restrict__ hc, const float* __restrict__ rc,
                         const float* __restrict__ xin, const float* __restrict__ emb,
                         const float* __restrict__ st1, const float* __restrict__ st2,
                         const float* __restrict__ g, const float* __restrict__ be,
                         const float* __restrict__ rg, const float* __restrict__ rbe,
                         const float* __restrict__ aS, long long P, int O, int Cin,
                         int KpadN, unsigned short* __restrict__ ybf, int TV, float invCnt) {
  int W = (KpadN > O) ? KpadN : O;
  long long total = P * W;
  long long i = (long long)blockIdx.x * blockDim.x + threadIdx.x;
  if (i >= total) return;
  int oc = (int)(i % W); long long p = i / W;
  if (oc >= O) { if (oc < KpadN) ybf[p * KpadN + oc] = 0; return; }
  float m = st1[oc * 2] * invCnt, q = st1[oc * 2 + 1] * invCnt;
  float rs = rsqrtf(q - m * m + BN_EPS);
  float h = g[oc] * (hc[p * O + oc] - m) * rs + be[oc];
  float r;
  if (rc) {
    float m2 = st2[oc * 2] * invCnt, q2 = st2[oc * 2 + 1] * invCnt;
    float rs2 = rsqrtf(q2 - m2 * m2 + BN_EPS);
    r = rg[oc] * (rc[p * O + oc] - m2) * rs2 + rbe[oc];
  } else {
    r = xin[p * Cin + oc];                   // Cin == O on no-res layers
  }
  h += r;
  float a = aS[0];
  h = (h >= 0.f) ? h : a * h;
  h += emb[(p / TV) * O + oc];
  hc[p * O + oc] = h;
  if (KpadN > 0) ybf[p * KpadN + oc] = f2bf(h);
}

// joint downsample: z[(n,t,w),c] = db[w] + sum_v y[(n,t,v),c] * dw[w,v]
__global__ void k_down(const float* __restrict__ y, float* __restrict__ z,
                       const float* __restrict__ dw, const float* __restrict__ db,
                       int N, int V, int W, int C) {
  long long total = (long long)N * 12 * W * C;
  long long i = (long long)blockIdx.x * blockDim.x + threadIdx.x;
  if (i >= total) return;
  int c = (int)(i % C); int w = (int)((i / C) % W);
  long long pp = i / ((long long)C * W);
  float s = db[w];
  const float* yp = y + (pp * V) * C + c;
  for (int v = 0; v < V; ++v) s += yp[(long long)v * C] * dw[w * V + v];
  z[i] = s;
}

// BN (per joint w) on down output; in-place fp32 + bf16 shadow (Kpad == C)
__global__ void k_dfinish(float* __restrict__ z, const float* __restrict__ st,
                          const float* __restrict__ g, const float* __restrict__ be,
                          long long total, int W, int C,
                          unsigned short* __restrict__ yb, float invCnt) {
  long long i = (long long)blockIdx.x * blockDim.x + threadIdx.x;
  if (i >= total) return;
  int w = (int)((i / C) % W);
  float m = st[w * 2] * invCnt, q = st[w * 2 + 1] * invCnt;
  float rs = rsqrtf(q - m * m + BN_EPS);
  float h = g[w] * (z[i] - m) * rs + be[w];
  z[i] = h;
  yb[i] = f2bf(h);
}

// (n,t,v,c) fp32 with C=6,V=10 -> flat (N x 736) bf16, k = c*120 + t*10 + v, zero pad
__global__ void k_repack(const float* __restrict__ y, unsigned short* __restrict__ fx, int N) {
  long long total = (long long)N * 736;
  long long i = (long long)blockIdx.x * blockDim.x + threadIdx.x;
  if (i >= total) return;
  int k = (int)(i % 736); int n = (int)(i / 736);
  if (k < 720) {
    int c = k / 120; int rr = k % 120; int t = rr / 10; int v = rr % 10;
    fx[i] = f2bf(y[(((long long)n * 12 + t) * 10 + v) * 6 + c]);
  } else fx[i] = 0;
}

// ---------------- host orchestration ----------------
#define ELW(kern, total, ...) do {                                              \
    long long _t = (total);                                                     \
    unsigned _g = (unsigned)((_t + 255) / 256);                                 \
    kern<<<dim3(_g), dim3(256), 0, stream>>>(__VA_ARGS__);                      \
  } while (0)

extern "C" void kernel_launch(void* const* d_in, const int* in_sizes, int n_in,
                              void* d_out, int out_size, void* d_ws, size_t ws_size,
                              hipStream_t stream) {
  (void)in_sizes; (void)n_in; (void)out_size; (void)ws_size;
  const int N = 4096;
  const float* X  = (const float*)d_in[0];
  const float* Tt = (const float*)d_in[1];

  // d_in leaf map (sorted-key pytree):
  // 0:x 1:t 2:d1.b 3:d1.be 4:d1.g 5:d1.w 6:d2.b 7:d2.be 8:d2.g 9:d2.w 10:final_b 11:final_w
  // per layer (base..): A,T,a,b,be,eb,ew,g,[res_b,res_be,res_g,res_w,]w
  struct LayerArg { int base, cin, cout, V, has_res; };
  static const LayerArg LS[7] = {
    {12, 2, 16, 17, 1}, {25, 16, 32, 17, 1}, {38, 32, 32, 17, 0},
    {47, 32, 64, 12, 1}, {60, 64, 64, 12, 0}, {69, 64, 128, 10, 1}, {82, 128, 6, 10, 1}
  };
  auto r32 = [](int x) { return (x + 31) & ~31; };
  auto r16 = [](int x) { return (x + 15) & ~15; };

  char* wp = (char*)d_ws;
  auto alloc = [&](size_t b) -> void* {
    void* r = (void*)wp; wp += (b + 255) & ~(size_t)255; return r;
  };
  const long long SZF = 62914560LL;  // max pixels*channels = (4096*12*10)*128
  float* F0 = (float*)alloc((size_t)SZF * 4);
  float* F1 = (float*)alloc((size_t)SZF * 4);
  unsigned short* XBF = (unsigned short*)alloc((size_t)SZF * 2);
  unsigned short* H2  = (unsigned short*)alloc((size_t)SZF * 2);
  float* SEMB = (float*)alloc((size_t)N * 12 * 4);
  float* EMBB = (float*)alloc((size_t)N * 128 * 4);
  float* PART = (float*)alloc((size_t)NB_STATS * 256 * 4);
  float* ST1  = (float*)alloc(256 * 4);
  float* ST2  = (float*)alloc(256 * 4);
  unsigned short* WB[7]; unsigned short* RWB[7];
  for (int l = 0; l < 7; ++l) {
    int Kp = r32(LS[l].cin), Mp = r16(LS[l].cout);
    WB[l]  = (unsigned short*)alloc((size_t)Mp * Kp * 2);
    RWB[l] = LS[l].has_res ? (unsigned short*)alloc((size_t)Mp * Kp * 2) : nullptr;
  }
  unsigned short* WF = (unsigned short*)alloc((size_t)512 * 736 * 2);
  unsigned short* FX = (unsigned short*)alloc((size_t)N * 736 * 2);

  // time embedding base + weight conversions
  ELW(k_silu, (long long)N * 12, Tt, SEMB, N * 12);
  for (int l = 0; l < 7; ++l) {
    int Kp = r32(LS[l].cin), Mp = r16(LS[l].cout);
    const float* w = (const float*)d_in[LS[l].base + (LS[l].has_res ? 12 : 8)];
    ELW(k_wconv, (long long)Mp * Kp, w, WB[l], LS[l].cout, LS[l].cin, Kp, (long long)Mp * Kp);
    if (LS[l].has_res) {
      const float* rw = (const float*)d_in[LS[l].base + 11];
      ELW(k_wconv, (long long)Mp * Kp, rw, RWB[l], LS[l].cout, LS[l].cin, Kp, (long long)Mp * Kp);
    }
  }
  ELW(k_wconv, (long long)512 * 736, (const float*)d_in[11], WF, 512, 720, 736,
      (long long)512 * 736);

  ELW(k_init, (long long)N * 12 * 17 * 32, X, F0, XBF, N);

  float* cur = F0; float* oth = F1;
  long long P = (long long)N * 12 * 17; int V = 17;

  auto do_down = [&](int bi, int W_, int C_) {
    const float* db  = (const float*)d_in[bi + 0];
    const float* dbe = (const float*)d_in[bi + 1];
    const float* dg  = (const float*)d_in[bi + 2];
    const float* dw  = (const float*)d_in[bi + 3];
    long long Pn = (long long)N * 12 * W_;
    ELW(k_down, Pn * C_, cur, oth, dw, db, N, V, W_, C_);
    k_stats<<<dim3(NB_STATS), dim3(256), 0, stream>>>(oth, Pn * C_, C_, W_, PART);
    k_sreduce<<<dim3(1), dim3(256), 0, stream>>>(PART, NB_STATS, W_, ST1);
    float invc = 1.0f / (float)((long long)N * 12 * C_);
    ELW(k_dfinish, Pn * C_, oth, ST1, dg, dbe, Pn * C_, W_, C_, XBF, invc);
    float* t2 = cur; cur = oth; oth = t2;
    V = W_; P = Pn;
  };

  for (int l = 0; l < 7; ++l) {
    if (l == 3) do_down(2, 12, 32);   // down1: 17 -> 12 joints, C=32
    if (l == 5) do_down(6, 10, 64);   // down2: 12 -> 10 joints, C=64
    const LayerArg& L = LS[l];
    int Kp = r32(L.cin);
    int Mt = (L.cout + 15) / 16;
    int KpN = (l == 0) ? 32 : (l == 1) ? 32 : (l == 3) ? 64 : (l == 5) ? 128 : 0;
    const float* Am = (const float*)d_in[L.base + 0];
    const float* Tm = (const float*)d_in[L.base + 1];
    const float* aS = (const float*)d_in[L.base + 2];
    const float* b  = (const float*)d_in[L.base + 3];
    const float* be = (const float*)d_in[L.base + 4];
    const float* eb = (const float*)d_in[L.base + 5];
    const float* ew = (const float*)d_in[L.base + 6];
    const float* g  = (const float*)d_in[L.base + 7];
    const float* rb  = L.has_res ? (const float*)d_in[L.base + 8]  : nullptr;
    const float* rbe = L.has_res ? (const float*)d_in[L.base + 9]  : nullptr;
    const float* rg  = L.has_res ? (const float*)d_in[L.base + 10] : nullptr;

    ELW(k_mix, (long long)N * 12 * Kp, cur, H2, Tm, Am, N, V, L.cin, Kp);
    {
      dim3 grid((unsigned)(P / 256), Mt);
      k_gemm<<<grid, dim3(128), 0, stream>>>(WB[l], H2, oth, b, L.cout, Kp, P);
    }
    if (L.has_res) {
      dim3 grid((unsigned)(P / 256), Mt);
      k_gemm<<<grid, dim3(128), 0, stream>>>(RWB[l], XBF, cur, rb, L.cout, Kp, P);
    }
    k_stats<<<dim3(NB_STATS), dim3(256), 0, stream>>>(oth, P * L.cout, 1, L.cout, PART);
    k_sreduce<<<dim3(1), dim3(256), 0, stream>>>(PART, NB_STATS, L.cout, ST1);
    if (L.has_res) {
      k_stats<<<dim3(NB_STATS), dim3(256), 0, stream>>>(cur, P * L.cout, 1, L.cout, PART);
      k_sreduce<<<dim3(1), dim3(256), 0, stream>>>(PART, NB_STATS, L.cout, ST2);
    }
    ELW(k_emb, (long long)N * L.cout, SEMB, ew, eb, EMBB, N, L.cout);
    float invc = 1.0f / (float)P;
    int Wt = (KpN > L.cout) ? KpN : L.cout;
    ELW(k_finish, P * Wt, oth,
        L.has_res ? cur : nullptr, L.has_res ? nullptr : cur,
        EMBB, ST1, ST2, g, be, rg, rbe, aS, P, L.cout, L.cin, KpN, XBF, 12 * V, invc);
    float* t2 = cur; cur = oth; oth = t2;
  }

  // final linear: (N x 720) @ (512 x 720)^T + b -> d_out (N x 512) fp32
  ELW(k_repack, (long long)N * 736, cur, FX, N);
  {
    dim3 grid((unsigned)(N / 256), 512 / 16);
    k_gemm<<<grid, dim3(128), 0, stream>>>(WF, FX, (float*)d_out,
                                           (const float*)d_in[10], 512, 736, (long long)N);
  }
}
